// MultiHeadAttention_7447473292228
// MI455X (gfx1250) — matrix operand order repaired and checked by reading, not compiled, not run on hardware
//
#include <hip/hip_runtime.h>
#include <hip/hip_bf16.h>

// ---------------------------------------------------------------------------
// MultiHeadAttention forward, CDNA5 (gfx1250), wave32.
//   bf16 WMMA (v_wmma_f32_16x16x32_bf16) for all matmuls,
//   GLOBAL_LOAD_ASYNC_TO_LDS_B128 (ASYNCcnt) for bf16 tile staging,
//   global_prefetch for next-tile warming.
//   B=4, S=2048, E=1024, H=16, Dh=64
// ---------------------------------------------------------------------------

typedef __attribute__((ext_vector_type(16))) __bf16 v16bf;
typedef __attribute__((ext_vector_type(8)))  __bf16 v8bf;
typedef __attribute__((ext_vector_type(8)))  float  v8f;

static constexpr int BATCH = 4;
static constexpr int SEQ   = 2048;
static constexpr int EMB   = 1024;
static constexpr int HEADS = 16;
static constexpr int DH    = 64;          // head dim
static constexpr int MROWS = BATCH * SEQ; // 8192

// LDS row strides (elements); multiples of 8 so every 8-element chunk is 16B
// aligned (80B / 144B pitches are multiples of 16B).
static constexpr int PK = 40;   // 32-wide K tiles (+8 pad)
static constexpr int PD = 72;   // 64-wide tiles   (+8 pad)

// --- CDNA5 async memory->LDS copy (no VGPR data path, tracked by ASYNCcnt) --
typedef __attribute__((address_space(3))) __bf16 lds_bf16_t;

__device__ __forceinline__ void async_copy16(__bf16* lds_dst, const __bf16* gsrc) {
    const unsigned loff = (unsigned)(size_t)(lds_bf16_t*)lds_dst;  // LDS byte offset
    asm volatile("global_load_async_to_lds_b128 %0, %1, off"
                 :: "v"(loff), "v"(gsrc)
                 : "memory");
}

__device__ __forceinline__ void wait_async0() {
#if __has_builtin(__builtin_amdgcn_s_wait_asynccnt)
    __builtin_amdgcn_s_wait_asynccnt(0);
#else
    asm volatile("s_wait_asynccnt 0" ::: "memory");
#endif
}

// --- WMMA fragment builders (layouts per CDNA5 ISA 7.12.2) -------------------
__device__ __forceinline__ v16bf make_a_frag(const __bf16* base, int row, int stride,
                                             int kchunk, int half) {
    // A 16x32 bf16: lane<16 -> K {0..7,16..23}; lane>=16 -> {8..15,24..31}
    const int k0 = kchunk + half * 8;
    union { v16bf v; v8bf h[2]; } u;
    u.h[0] = *(const v8bf*)&base[row * stride + k0];
    u.h[1] = *(const v8bf*)&base[row * stride + k0 + 16];
    return u.v;
}

__device__ __forceinline__ v16bf make_b_frag(const __bf16* base, int col, int stride,
                                             int kchunk, int half) {
    // B 32x16 bf16: lane<16 -> column lane, K {0..7,16..23}; lane>=16 -> {8..15,24..31}
    const int kb = kchunk + half * 8;
    union { v16bf v; v8bf h[2]; } u;
    u.h[0] = *(const v8bf*)&base[col * stride + kb];
    u.h[1] = *(const v8bf*)&base[col * stride + kb + 16];
    return u.v;
}

// ---------------------------------------------------------------------------
// Kernel 1: projection  (X @ W^T) -> bf16, head layout.
//   VTRANS=false: dst[b,h,s,dh]   (Q, K)
//   VTRANS=true : dst[b,h,dh,s]   (V, pre-transposed for the PV matmul)
//   X: [8192,1024] f32, W: [1024,1024] f32.  256 thr (8 waves), 128x64, BK=32
// ---------------------------------------------------------------------------
template <bool VTRANS>
__global__ __launch_bounds__(256) void gemm_qkv_kernel(
    const float* __restrict__ X, const float* __restrict__ W,
    __bf16* __restrict__ dst) {
    __shared__ __bf16 As[128 * PK];
    __shared__ __bf16 Bs[64 * PK];

    const int tid  = threadIdx.x;
    const int wave = tid >> 5;
    const int lane = tid & 31;
    const int half = lane >> 4;
    const int ln   = lane & 15;
    const int m0   = blockIdx.x * 128;
    const int n0   = blockIdx.y * 64;

    v8f acc[4] = {};

    for (int kk = 0; kk < EMB; kk += 32) {
        __syncthreads();
        // A tile: 128x32 f32 -> bf16 (1024 float4, 4/thread)
#pragma unroll
        for (int j = 0; j < 4; ++j) {
            const int idx = tid + 256 * j;
            const int row = idx >> 3;
            const int kq  = (idx & 7) * 4;
            const float* gp = &X[(size_t)(m0 + row) * EMB + kk + kq];
            const float4 f = *(const float4*)gp;
            if (kk + 32 < EMB) __builtin_prefetch(gp + 32, 0, 0);
            __bf16* p = &As[row * PK + kq];
            p[0] = (__bf16)f.x; p[1] = (__bf16)f.y; p[2] = (__bf16)f.z; p[3] = (__bf16)f.w;
        }
        // B tile: 64x32 f32 -> bf16 (512 float4, 2/thread)
#pragma unroll
        for (int j = 0; j < 2; ++j) {
            const int idx = tid + 256 * j;
            const int row = idx >> 3;
            const int kq  = (idx & 7) * 4;
            const float* gp = &W[(size_t)(n0 + row) * EMB + kk + kq];
            const float4 f = *(const float4*)gp;
            if (kk + 32 < EMB) __builtin_prefetch(gp + 32, 0, 0);
            __bf16* p = &Bs[row * PK + kq];
            p[0] = (__bf16)f.x; p[1] = (__bf16)f.y; p[2] = (__bf16)f.z; p[3] = (__bf16)f.w;
        }
        __syncthreads();

        const v16bf a = make_a_frag(As, wave * 16 + ln, PK, 0, half);
#pragma unroll
        for (int tn = 0; tn < 4; ++tn) {
            const v16bf bfr = make_b_frag(Bs, tn * 16 + ln, PK, 0, half);
            acc[tn] = __builtin_amdgcn_wmma_f32_16x16x32_bf16(
                false, a, false, bfr, (short)0, acc[tn], false, false);
        }
    }

    // C layout: row = i + 8*half (VGPR i), col = ln
    const int rbase = wave * 16 + half * 8;
#pragma unroll
    for (int tn = 0; tn < 4; ++tn) {
        const int n  = n0 + tn * 16 + ln;
        const int h  = n >> 6;        // / DH
        const int dh = n & (DH - 1);
#pragma unroll
        for (int i = 0; i < 8; ++i) {
            const int m = m0 + rbase + i;
            const int b = m >> 11;    // / SEQ
            const int s = m & (SEQ - 1);
            const __bf16 v = (__bf16)acc[tn][i];
            if (VTRANS)
                dst[(((size_t)b * HEADS + h) * DH + dh) * SEQ + s] = v;
            else
                dst[(((size_t)b * HEADS + h) * SEQ + s) * DH + dh] = v;
        }
    }
}

// ---------------------------------------------------------------------------
// Kernel 2: flash attention over one (b,h); 64-query tile / block.
//   128 threads (4 waves); each wave owns a 16-row query strip.
//   Qh/Kh: [b,h,s,dh] bf16.  Vh: [b,h,dh,s] bf16 (pre-transposed).
//   Tiles staged with GLOBAL_LOAD_ASYNC_TO_LDS_B128.
// ---------------------------------------------------------------------------
__global__ __launch_bounds__(128) void attn_kernel(
    const __bf16* __restrict__ Qh, const __bf16* __restrict__ Kh,
    const __bf16* __restrict__ Vh, const unsigned char* __restrict__ mask,
    __bf16* __restrict__ ctx) {
    __shared__ __bf16 Qt[64 * PD];   // [q][dh]
    __shared__ __bf16 Kt[64 * PD];   // [key][dh]
    __shared__ __bf16 Vt[64 * PD];   // [dh][key]
    __shared__ __bf16 Pt[64 * PD];   // [q][key]

    const int tid  = threadIdx.x;
    const int wave = tid >> 5;
    const int lane = tid & 31;
    const int half = lane >> 4;
    const int ln   = lane & 15;
    const int q0   = blockIdx.x * 64;
    const int bh   = blockIdx.y;
    const int b    = bh / HEADS;
    const int h    = bh % HEADS;

    const __bf16* Qp = Qh + ((size_t)bh * SEQ + q0) * DH;
    const __bf16* Kp = Kh + (size_t)bh * SEQ * DH;
    const __bf16* Vp = Vh + (size_t)bh * DH * SEQ;   // [dh][s]

    // Q tile once (64x64 bf16; 512 16B chunks, 4/thread), async to LDS
#pragma unroll
    for (int j = 0; j < 4; ++j) {
        const int idx = tid + 128 * j;
        const int row = idx >> 3;
        const int c   = (idx & 7) * 8;
        async_copy16(&Qt[row * PD + c], &Qp[(size_t)row * DH + c]);
    }

    float mrow[8], lrow[8];
    v8f Oacc[4] = {};
#pragma unroll
    for (int i = 0; i < 8; ++i) { mrow[i] = -1e30f; lrow[i] = 0.f; }

    const float scale = 0.125f;   // 1/sqrt(64)

    for (int kt = 0; kt < SEQ; kt += 64) {
        __syncthreads();   // all waves done reading previous Kt/Vt
        // K tile [key][dh] + V tile [dh][key], both contiguous 16B chunks
#pragma unroll
        for (int j = 0; j < 4; ++j) {
            const int idx = tid + 128 * j;
            const int row = idx >> 3;
            const int c   = (idx & 7) * 8;
            async_copy16(&Kt[row * PD + c], &Kp[(size_t)(kt + row) * DH + c]);
            async_copy16(&Vt[row * PD + c], &Vp[(size_t)row * SEQ + kt + c]);
            if (kt + 64 < SEQ) {
                __builtin_prefetch(&Kp[(size_t)(kt + 64 + row) * DH + c], 0, 0);
                __builtin_prefetch(&Vp[(size_t)row * SEQ + kt + 64 + c], 0, 0);
            }
        }
        wait_async0();
        __syncthreads();

        // scores = Q @ K^T  (16 rows x 64 keys per wave)
        v8f sc[4] = {};
#pragma unroll
        for (int c = 0; c < 2; ++c) {
            const v16bf a = make_a_frag(Qt, wave * 16 + ln, PD, c * 32, half);
#pragma unroll
            for (int tn = 0; tn < 4; ++tn) {
                const v16bf bfr = make_b_frag(Kt, tn * 16 + ln, PD, c * 32, half);
                sc[tn] = __builtin_amdgcn_wmma_f32_16x16x32_bf16(
                    false, a, false, bfr, (short)0, sc[tn], false, false);
            }
        }

        // key-validity mask for this lane's columns
        float kmask[4];
#pragma unroll
        for (int tn = 0; tn < 4; ++tn) {
            const int key = kt + tn * 16 + ln;
            kmask[tn] = mask[(size_t)b * SEQ + key] ? 0.f : -1e30f;
        }

        // online softmax per row (rows live in 16-lane halves; xor<16 stays in half)
        const int prow = wave * 16 + half * 8;
#pragma unroll
        for (int i = 0; i < 8; ++i) {
            float rmax = -1e30f;
#pragma unroll
            for (int tn = 0; tn < 4; ++tn) {
                const float sv = sc[tn][i] * scale + kmask[tn];
                sc[tn][i] = sv;
                rmax = fmaxf(rmax, sv);
            }
            for (int off = 1; off < 16; off <<= 1)
                rmax = fmaxf(rmax, __shfl_xor(rmax, off, 32));
            const float mnew = fmaxf(mrow[i], rmax);
            const float f = __expf(mrow[i] - mnew);
            mrow[i] = mnew;
            float rsum = 0.f;
#pragma unroll
            for (int tn = 0; tn < 4; ++tn) {
                const float p = __expf(sc[tn][i] - mnew);
                rsum += p;
                Pt[(prow + i) * PD + tn * 16 + ln] = (__bf16)p;
            }
            for (int off = 1; off < 16; off <<= 1)
                rsum += __shfl_xor(rsum, off, 32);
            lrow[i] = lrow[i] * f + rsum;
#pragma unroll
            for (int tn = 0; tn < 4; ++tn) Oacc[tn][i] *= f;
        }
        __syncthreads();

        // O += P @ V   (B frags from [dh][key] tile: contiguous over key dim)
#pragma unroll
        for (int c = 0; c < 2; ++c) {
            const v16bf a = make_a_frag(Pt, wave * 16 + ln, PD, c * 32, half);
#pragma unroll
            for (int tn = 0; tn < 4; ++tn) {
                const v16bf bfr = make_b_frag(Vt, tn * 16 + ln, PD, c * 32, half);
                Oacc[tn] = __builtin_amdgcn_wmma_f32_16x16x32_bf16(
                    false, a, false, bfr, (short)0, Oacc[tn], false, false);
            }
        }
    }

    // normalize, write ctx[b, s, h*DH + dh] bf16
#pragma unroll
    for (int i = 0; i < 8; ++i) {
        const float inv = 1.f / fmaxf(lrow[i], 1e-30f);
        const int s = q0 + wave * 16 + half * 8 + i;
#pragma unroll
        for (int tn = 0; tn < 4; ++tn) {
            const int dh = tn * 16 + ln;
            ctx[((size_t)b * SEQ + s) * EMB + h * DH + dh] = (__bf16)(Oacc[tn][i] * inv);
        }
    }
}

// ---------------------------------------------------------------------------
// Kernel 3: out = ctx(bf16) @ Wc^T + bc  -> f32
//   A tile staged async (bf16, no conversion); B tile f32->bf16 via VALU.
// ---------------------------------------------------------------------------
__global__ __launch_bounds__(256) void gemm_out_kernel(
    const __bf16* __restrict__ X, const float* __restrict__ W,
    const float* __restrict__ bias, float* __restrict__ out) {
    __shared__ __bf16 As[128 * PK];
    __shared__ __bf16 Bs[64 * PK];

    const int tid  = threadIdx.x;
    const int wave = tid >> 5;
    const int lane = tid & 31;
    const int half = lane >> 4;
    const int ln   = lane & 15;
    const int m0   = blockIdx.x * 128;
    const int n0   = blockIdx.y * 64;

    v8f acc[4] = {};

    for (int kk = 0; kk < EMB; kk += 32) {
        __syncthreads();
        // A tile: 128x32 bf16 (512 16B chunks, 2/thread) async to LDS
#pragma unroll
        for (int j = 0; j < 2; ++j) {
            const int idx = tid + 256 * j;
            const int row = idx >> 2;
            const int kq  = (idx & 3) * 8;
            const __bf16* gp = &X[(size_t)(m0 + row) * EMB + kk + kq];
            async_copy16(&As[row * PK + kq], gp);
            if (kk + 32 < EMB) __builtin_prefetch(gp + 32, 0, 0);
        }
        // B tile: 64x32 f32 -> bf16
#pragma unroll
        for (int j = 0; j < 2; ++j) {
            const int idx = tid + 256 * j;
            const int row = idx >> 3;
            const int kq  = (idx & 7) * 4;
            const float* gp = &W[(size_t)(n0 + row) * EMB + kk + kq];
            const float4 f = *(const float4*)gp;
            if (kk + 32 < EMB) __builtin_prefetch(gp + 32, 0, 0);
            __bf16* p = &Bs[row * PK + kq];
            p[0] = (__bf16)f.x; p[1] = (__bf16)f.y; p[2] = (__bf16)f.z; p[3] = (__bf16)f.w;
        }
        wait_async0();
        __syncthreads();

        const v16bf a = make_a_frag(As, wave * 16 + ln, PK, 0, half);
#pragma unroll
        for (int tn = 0; tn < 4; ++tn) {
            const v16bf bfr = make_b_frag(Bs, tn * 16 + ln, PK, 0, half);
            acc[tn] = __builtin_amdgcn_wmma_f32_16x16x32_bf16(
                false, a, false, bfr, (short)0, acc[tn], false, false);
        }
    }

    const int rbase = wave * 16 + half * 8;
#pragma unroll
    for (int tn = 0; tn < 4; ++tn) {
        const int n = n0 + tn * 16 + ln;
        const float bv = bias[n];
#pragma unroll
        for (int i = 0; i < 8; ++i) {
            const int m = m0 + rbase + i;
            out[(size_t)m * EMB + n] = acc[tn][i] + bv;
        }
    }
}

// ---------------------------------------------------------------------------
extern "C" void kernel_launch(void* const* d_in, const int* in_sizes, int n_in,
                              void* d_out, int out_size, void* d_ws, size_t ws_size,
                              hipStream_t stream) {
    (void)in_sizes; (void)n_in; (void)out_size; (void)ws_size;
    const float* query = (const float*)d_in[0];
    const float* key   = (const float*)d_in[1];
    const float* value = (const float*)d_in[2];
    const unsigned char* mask = (const unsigned char*)d_in[3];  // bool [B,S,1]
    const float* Wq = (const float*)d_in[4];
    const float* Wk = (const float*)d_in[5];
    const float* Wv = (const float*)d_in[6];
    const float* Wc = (const float*)d_in[7];
    const float* bc = (const float*)d_in[8];
    float* out = (float*)d_out;

    // workspace: Qh | Kh | Vh | Ctx, each 16 MiB of bf16
    char* ws = (char*)d_ws;
    const size_t headBytes = (size_t)BATCH * HEADS * SEQ * DH * sizeof(__bf16);
    __bf16* Qh  = (__bf16*)(ws);
    __bf16* Kh  = (__bf16*)(ws + headBytes);
    __bf16* Vh  = (__bf16*)(ws + 2 * headBytes);   // [b,h,dh,s]
    __bf16* Ctx = (__bf16*)(ws + 3 * headBytes);

    dim3 gridP(MROWS / 128, EMB / 64);   // 64 x 16
    gemm_qkv_kernel<false><<<gridP, 256, 0, stream>>>(query, Wq, Qh);
    gemm_qkv_kernel<false><<<gridP, 256, 0, stream>>>(key,   Wk, Kh);
    gemm_qkv_kernel<true ><<<gridP, 256, 0, stream>>>(value, Wv, Vh);

    dim3 gridA(SEQ / 64, BATCH * HEADS); // 32 x 64
    attn_kernel<<<gridA, 128, 0, stream>>>(Qh, Kh, Vh, mask, Ctx);

    gemm_out_kernel<<<gridP, 256, 0, stream>>>(Ctx, Wc, bc, out);
}
